// SemanticLossModule_27290222199325
// MI455X (gfx1250) — matrix operand order repaired and checked by reading, not compiled, not run on hardware
//
#include <hip/hip_runtime.h>
#include <hip/hip_bf16.h>
#include <math.h>

typedef __attribute__((ext_vector_type(16))) _Float16 v16h;
typedef __attribute__((ext_vector_type(8)))  float    v8f;

#define D_IN    784
#define D_PAD   800     // 25 chunks of K=32
#define H1      512
#define H2      128
#define NC      10
#define NCP     16
#define TILE_M  16
#define MACRO_M 64      // rows per workgroup
#define MT      4       // m-tiles per workgroup (MACRO_M / TILE_M)
#define NTHREADS 128    // 4 waves

// ---- WMMA fragment loaders (16-bit, contiguous k = 16*(lane/16) + elem for A and B, wave32) ----

// A (16x32, MxK) from a row-major [16][ld] f16 tile.
// lanes 0-15: M=lane, elems 0-15 -> K=k0+0..15
// lanes 16-31: M=lane-16, elems 0-15 -> K=k0+16..31
__device__ __forceinline__ v16h load_A_rm(const _Float16* base, int ld, int k0, int lane) {
  const _Float16* p = base + (lane & 15) * ld + k0 + ((lane >> 4) << 4);
  v16h a;
#pragma unroll
  for (int j = 0; j < 8; ++j) a[j] = p[j];
#pragma unroll
  for (int j = 0; j < 8; ++j) a[8 + j] = p[8 + j];
  return a;
}

// B (32x16, KxN) where memory is row-major W[N][K] (i.e. B[k][n] = W[n][k]).
// lane holds column N = n0 + (lane&15); elem j -> K = k0 + 16*(lane>>4) + j
// => one contiguous 32-byte run per lane.
__device__ __forceinline__ v16h load_B_rm(const _Float16* W, int ld, int n0, int k0, int lane) {
  const _Float16* p = W + (size_t)(n0 + (lane & 15)) * ld + k0 + ((lane >> 4) << 4);
  v16h b;
#pragma unroll
  for (int j = 0; j < 16; ++j) b[j] = p[j];
  return b;
}

// ---- weight conversion / padding + accumulator init ----
__global__ void prep_kernel(const float* __restrict__ W1, const float* __restrict__ W2,
                            const float* __restrict__ W3,
                            _Float16* __restrict__ W1h, _Float16* __restrict__ W2h,
                            _Float16* __restrict__ W3h, float* __restrict__ bce_acc) {
  const int i = blockIdx.x * blockDim.x + threadIdx.x;
  if (i == 0) bce_acc[0] = 0.f;
  if (i < H1 * D_PAD) {
    const int n = i / D_PAD, k = i % D_PAD;
    W1h[i] = (k < D_IN) ? (_Float16)W1[n * D_IN + k] : (_Float16)0.f;
  }
  if (i < H2 * H1) W2h[i] = (_Float16)W2[i];
  if (i < NCP * H2) {
    const int n = i / H2, k = i % H2;
    W3h[i] = (n < NC) ? (_Float16)W3[n * H2 + k] : (_Float16)0.f;
  }
}

// ---- fully fused MLP over a 64-row macro-tile per workgroup ----
// N split across waves; each B fragment is reused for 4 m-tiles (register blocking).
__global__ __launch_bounds__(NTHREADS) void mlp_fused_kernel(
    const float* __restrict__ x, const int* __restrict__ y,
    const _Float16* __restrict__ W1h, const float* __restrict__ b1,
    const _Float16* __restrict__ W2h, const float* __restrict__ b2,
    const _Float16* __restrict__ W3h, const float* __restrict__ b3,
    float* __restrict__ out_logits, float* __restrict__ bce_acc,
    int nrows, int mode) {
  extern __shared__ _Float16 smem[];
  _Float16* xw  = smem;                                  // [64][800] f16
  _Float16* h1w = smem + MACRO_M * D_PAD;                // [64][512] f16
  _Float16* h2w = smem + MACRO_M * (D_PAD + H1);         // [64][128] f16

  const int tid  = threadIdx.x;
  const int wave = tid >> 5;
  const int lane = tid & 31;
  const int row0 = blockIdx.x * MACRO_M;
  if (row0 >= nrows) return;   // block-uniform

  const int c    = lane & 15;          // fragment column index
  const int mrow = (lane >> 4) << 3;   // 0 or 8: C/D row base for this half-wave

  // ---- stage x macro-tile as f16 into LDS (f32 float4 loads), zero K-pad ----
  for (int r = 0; r < MACRO_M; ++r) {
    const float* xr = x + (size_t)(row0 + r) * D_IN;
    for (int k = tid * 4; k < D_PAD; k += NTHREADS * 4) {
      float4 v;
      if (k + 3 < D_IN) {
        v = *(const float4*)(xr + k);
      } else {
        v.x = (k + 0 < D_IN) ? xr[k + 0] : 0.f;
        v.y = (k + 1 < D_IN) ? xr[k + 1] : 0.f;
        v.z = (k + 2 < D_IN) ? xr[k + 2] : 0.f;
        v.w = (k + 3 < D_IN) ? xr[k + 3] : 0.f;
      }
      _Float16* d = xw + r * D_PAD + k;
      d[0] = (_Float16)v.x; d[1] = (_Float16)v.y;
      d[2] = (_Float16)v.z; d[3] = (_Float16)v.w;
    }
  }
  __syncthreads();

  // ---- layer 1: h1 = relu(x @ W1^T + b1); N=512, each wave owns 8 n-tiles ----
  for (int nti = 0; nti < (H1 / 16) / 4; ++nti) {
    const int n0 = (wave * 8 + nti) * 16;
    const float bias = b1[n0 + c];
    v8f acc[MT];
#pragma unroll
    for (int m = 0; m < MT; ++m)
#pragma unroll
      for (int r = 0; r < 8; ++r) acc[m][r] = bias;

    const _Float16* brow = W1h + (size_t)(n0 + c) * D_PAD + ((lane >> 4) << 4);
    for (int k0 = 0; k0 < D_PAD; k0 += 32) {
      __builtin_prefetch(brow + k0 + 32, 0, 1);            // global_prefetch_b8
      v16h b = load_B_rm(W1h, D_PAD, n0, k0, lane);
#pragma unroll
      for (int m = 0; m < MT; ++m) {
        v16h a = load_A_rm(xw + m * TILE_M * D_PAD, D_PAD, k0, lane);
        acc[m] = __builtin_amdgcn_wmma_f32_16x16x32_f16(false, a, false, b, (short)0,
                                                        acc[m], false, false);
      }
    }
#pragma unroll
    for (int m = 0; m < MT; ++m)
#pragma unroll
      for (int r = 0; r < 8; ++r)
        h1w[(m * TILE_M + r + mrow) * H1 + n0 + c] = (_Float16)fmaxf(acc[m][r], 0.f);
  }
  __syncthreads();

  // ---- layer 2: h2 = relu(h1 @ W2^T + b2); N=128, each wave owns 2 n-tiles ----
  for (int nti = 0; nti < (H2 / 16) / 4; ++nti) {
    const int n0 = (wave * 2 + nti) * 16;
    const float bias = b2[n0 + c];
    v8f acc[MT];
#pragma unroll
    for (int m = 0; m < MT; ++m)
#pragma unroll
      for (int r = 0; r < 8; ++r) acc[m][r] = bias;

    for (int k0 = 0; k0 < H1; k0 += 32) {
      v16h b = load_B_rm(W2h, H1, n0, k0, lane);
#pragma unroll
      for (int m = 0; m < MT; ++m) {
        v16h a = load_A_rm(h1w + m * TILE_M * H1, H1, k0, lane);
        acc[m] = __builtin_amdgcn_wmma_f32_16x16x32_f16(false, a, false, b, (short)0,
                                                        acc[m], false, false);
      }
    }
#pragma unroll
    for (int m = 0; m < MT; ++m)
#pragma unroll
      for (int r = 0; r < 8; ++r)
        h2w[(m * TILE_M + r + mrow) * H2 + n0 + c] = (_Float16)fmaxf(acc[m][r], 0.f);
  }
  __syncthreads();

  // ---- layer 3: logits = h2 @ W3^T + b3; single n-tile, waves split m ----
  {
    const _Float16* h2t = h2w + wave * TILE_M * H2;
    const int mbase = row0 + wave * TILE_M;
    const float bias = (c < NC) ? b3[c] : 0.f;
    v8f acc;
#pragma unroll
    for (int r = 0; r < 8; ++r) acc[r] = bias;
    for (int k0 = 0; k0 < H2; k0 += 32) {
      v16h a = load_A_rm(h2t, H2, k0, lane);
      v16h b = load_B_rm(W3h, H2, 0, k0, lane);
      acc = __builtin_amdgcn_wmma_f32_16x16x32_f16(false, a, false, b, (short)0, acc,
                                                   false, false);
    }

    if (mode == 0) {
      float local = 0.f;
      if (c < NC) {
#pragma unroll
        for (int r = 0; r < 8; ++r) {
          const int m = mbase + r + mrow;
          const float l = acc[r];
          const float t = (y[m] == c) ? 1.f : 0.f;
          local += fmaxf(l, 0.f) - l * t + log1pf(expf(-fabsf(l)));
        }
      }
#pragma unroll
      for (int off = 16; off >= 1; off >>= 1) local += __shfl_xor(local, off, 32);
      if (lane == 0) atomicAdd(bce_acc, local);
    } else {
      if (c < NC) {
#pragma unroll
        for (int r = 0; r < 8; ++r) {
          const int m = mbase + r + mrow;
          out_logits[(size_t)m * NC + c] = acc[r];
        }
      }
    }
  }
}

// ---- semantic loss + finalize ----
// log_sigmoid(o) - log_sigmoid(-o) == o, so
// log_s[c] = logsumexp_i(o[i][c]) + sum_i log_sigmoid(-o[i][c])
__global__ __launch_bounds__(320) void semantic_kernel(const float* __restrict__ o, int Bu,
                                                       const float* __restrict__ bce_acc,
                                                       float inv_count,
                                                       float* __restrict__ out) {
  __shared__ float sl_parts[NC];
  const int wave = threadIdx.x >> 5;
  const int lane = threadIdx.x & 31;
  if (wave < NC) {
    const int cc = wave;
    float colsum = 0.f, mx = -INFINITY;
    for (int i = lane; i < Bu; i += 32) {
      const float v = o[(size_t)i * NC + cc];
      colsum += (v > 0.f) ? (-v - log1pf(expf(-v))) : (-log1pf(expf(v)));
      mx = fmaxf(mx, v);
    }
#pragma unroll
    for (int off = 16; off >= 1; off >>= 1) {
      colsum += __shfl_xor(colsum, off, 32);
      mx = fmaxf(mx, __shfl_xor(mx, off, 32));
    }
    float se = 0.f;
    for (int i = lane; i < Bu; i += 32) se += expf(o[(size_t)i * NC + cc] - mx);
#pragma unroll
    for (int off = 16; off >= 1; off >>= 1) se += __shfl_xor(se, off, 32);
    if (lane == 0) sl_parts[cc] = -(logf(se) + mx + colsum);
  }
  __syncthreads();
  if (threadIdx.x == 0) {
    float s = 0.f;
    for (int k = 0; k < NC; ++k) s += sl_parts[k];
    out[1] = s / (float)NC;
    out[0] = bce_acc[0] * inv_count;
  }
}

extern "C" void kernel_launch(void* const* d_in, const int* in_sizes, int n_in,
                              void* d_out, int out_size, void* d_ws, size_t ws_size,
                              hipStream_t stream) {
  const float* x  = (const float*)d_in[0];
  const int*   y  = (const int*)d_in[1];
  const float* xu = (const float*)d_in[2];
  const float* W1 = (const float*)d_in[3];
  const float* b1 = (const float*)d_in[4];
  const float* W2 = (const float*)d_in[5];
  const float* b2 = (const float*)d_in[6];
  const float* W3 = (const float*)d_in[7];
  const float* b3 = (const float*)d_in[8];

  const int B  = in_sizes[0] / D_IN;   // 65536
  const int Bu = in_sizes[2] / D_IN;   // 4096

  char* ws = (char*)d_ws;
  float*    bce_acc = (float*)ws;                                            // 256 B slot
  _Float16* W1h = (_Float16*)(ws + 256);                                     // 512*800*2
  _Float16* W2h = (_Float16*)(ws + 256 + (size_t)H1 * D_PAD * 2);            // 128*512*2
  _Float16* W3h = (_Float16*)(ws + 256 + (size_t)H1 * D_PAD * 2
                                      + (size_t)H2 * H1 * 2);                // 16*128*2
  float*    o_u = (float*)(ws + 256 + (size_t)H1 * D_PAD * 2
                                    + (size_t)H2 * H1 * 2
                                    + (size_t)NCP * H2 * 2);                 // Bu*10*4

  prep_kernel<<<(H1 * D_PAD + 255) / 256, 256, 0, stream>>>(W1, W2, W3, W1h, W2h, W3h,
                                                            bce_acc);

  const size_t shmem = (size_t)MACRO_M * (D_PAD + H1 + H2) * sizeof(_Float16);
  const int grid_l = (B + MACRO_M - 1) / MACRO_M;
  mlp_fused_kernel<<<grid_l, NTHREADS, shmem, stream>>>(x, y, W1h, b1, W2h, b2, W3h, b3,
                                                        nullptr, bce_acc, B, 0);
  const int grid_u = (Bu + MACRO_M - 1) / MACRO_M;
  mlp_fused_kernel<<<grid_u, NTHREADS, shmem, stream>>>(xu, nullptr, W1h, b1, W2h, b2,
                                                        W3h, b3, o_u, nullptr, Bu, 1);
  semantic_kernel<<<1, 320, 0, stream>>>(o_u, Bu, bce_acc, 1.f / ((float)B * (float)NC),
                                         (float*)d_out);
}
